// RadioNaclipAttn_61959198212470
// MI455X (gfx1250) — compile-verified, matrix operand order checked by reading, not run on hardware
//
#include <hip/hip_runtime.h>
#include <hip/hip_bf16.h>

// ---------------------------------------------------------------------------
// RadioNaclipAttn on MI455X (gfx1250, wave32, WMMA).
//   x[8,1024,768] -> kv proj (skip q!) -> flash attn (scores = k.k^T*0.125 +
//   bias, softmax, @v) -> out proj.  All matmuls via v_wmma_f32_16x16x32_bf16.
// Workspace layout (bf16): K[8,12,1024,64] | V[8,12,1024,64] | A[8,1024,768]
//   = 3 * 12.58 MB = ~37.7 MB.
// ---------------------------------------------------------------------------

typedef __attribute__((ext_vector_type(16))) __bf16 v16bf;
typedef __attribute__((ext_vector_type(8)))  float  v8f;

#define Bsz 8
#define Nseq 1024
#define Cdim 768
#define Hn 12
#define HD 64

// CDNA5 async global->LDS path (ASYNCcnt).  The builtin takes int4* operands
// (signature revealed by the compiler diagnostic): (src, dst, offset, cpol).
#if __has_builtin(__builtin_amdgcn_global_load_async_to_lds_b128) && \
    __has_builtin(__builtin_amdgcn_s_wait_asynccnt)
#define HAVE_ASYNC_LDS 1
typedef int async_vec __attribute__((vector_size(16)));  // int4, 16 bytes
#endif

__device__ __forceinline__ v8f wmma_bf16(v16bf a, v16bf b, v8f c) {
  return __builtin_amdgcn_wmma_f32_16x16x32_bf16(false, a, false, b,
                                                 (short)0, c, false, false);
}

// load 16 contiguous f32, convert to bf16 fragment (lane supplies base ptr)
__device__ __forceinline__ v16bf cvt16(const float* __restrict__ p) {
  v16bf r;
  const float4* q = (const float4*)p;
#pragma unroll
  for (int i = 0; i < 4; ++i) {
    float4 f = q[i];
    r[4 * i + 0] = (__bf16)f.x;
    r[4 * i + 1] = (__bf16)f.y;
    r[4 * i + 2] = (__bf16)f.z;
    r[4 * i + 3] = (__bf16)f.w;
  }
  return r;
}

// ---------------------------------------------------------------------------
// Kernel 1: K/V projection.  out[m, j] = x[m,:] . Wqkv[768+j, :] + bqkv[768+j]
// M = 8192, Ncols = 1536 (k then v), K = 768.  Block tile 128x128, 8 waves,
// each wave 32(M) x 64(N) = 8 accumulator tiles.
// ---------------------------------------------------------------------------
__global__ __launch_bounds__(256) void kv_proj_kernel(
    const float* __restrict__ x, const float* __restrict__ Wqkv,
    const float* __restrict__ bqkv, __bf16* __restrict__ kws,
    __bf16* __restrict__ vws) {
  const int lane = threadIdx.x & 31, w = threadIdx.x >> 5;
  const int lr = lane & 15, lh = lane >> 4;
  const int m0 = blockIdx.x * 128 + (w & 3) * 32;
  const int n0 = blockIdx.y * 128 + (w >> 2) * 64;

  v8f acc[2][4];
#pragma unroll
  for (int i = 0; i < 2; ++i)
#pragma unroll
    for (int t = 0; t < 4; ++t) acc[i][t] = (v8f){};

  for (int k0 = 0; k0 < Cdim; k0 += 32) {
    __builtin_prefetch(&x[(size_t)(m0 + lr) * Cdim + k0 + 256], 0, 0);
    v16bf a[2], bm[4];
#pragma unroll
    for (int i = 0; i < 2; ++i)
      a[i] = cvt16(&x[(size_t)(m0 + i * 16 + lr) * Cdim + k0 + lh * 16]);
#pragma unroll
    for (int t = 0; t < 4; ++t)
      bm[t] = cvt16(
          &Wqkv[(size_t)(Cdim + n0 + t * 16 + lr) * Cdim + k0 + lh * 16]);
#pragma unroll
    for (int i = 0; i < 2; ++i)
#pragma unroll
      for (int t = 0; t < 4; ++t) acc[i][t] = wmma_bf16(a[i], bm[t], acc[i][t]);
  }

#pragma unroll
  for (int i = 0; i < 2; ++i)
#pragma unroll
    for (int t = 0; t < 4; ++t)
#pragma unroll
      for (int r = 0; r < 8; ++r) {
        int m = m0 + i * 16 + r + 8 * lh;           // global row
        int j = n0 + t * 16 + lr;                   // col in [0,1536)
        float val = acc[i][t][r] + bqkv[Cdim + j];
        int b = m >> 10, n = m & 1023;
        if (j < Cdim) {
          int h = j >> 6, d = j & 63;
          kws[(((size_t)(b * Hn + h)) * Nseq + n) * HD + d] = (__bf16)val;
        } else {
          int jj = j - Cdim;
          int h = jj >> 6, d = jj & 63;
          vws[(((size_t)(b * Hn + h)) * Nseq + n) * HD + d] = (__bf16)val;
        }
      }
}

// ---------------------------------------------------------------------------
// Kernel 2: fused attention per (b, h, 64-row block).  128 threads = 4 waves,
// each wave owns 16 query rows.  Flash softmax over j in steps of 32.
// V tile staged row-major into LDS via async global->LDS copies (CDNA5 path,
// ASYNCcnt); the j<->d transpose happens at B-fragment build time.
// ---------------------------------------------------------------------------
__global__ __launch_bounds__(128) void attn_kernel(
    const __bf16* __restrict__ kws, const __bf16* __restrict__ vws,
    const float* __restrict__ bias, __bf16* __restrict__ aws) {
  const int lane = threadIdx.x & 31, w = threadIdx.x >> 5;
  const int lr = lane & 15, lh = lane >> 4;
  const int rb = blockIdx.x & 15;
  const int h = (blockIdx.x >> 4) % Hn;
  const int b = blockIdx.x / (16 * Hn);
  const int m0 = rb * 64 + w * 16;

  const __bf16* Kb = kws + ((size_t)(b * Hn + h)) * Nseq * HD;
  const __bf16* Vb = vws + ((size_t)(b * Hn + h)) * Nseq * HD;

  __shared__ __bf16 vRow[32][HD];      // V rows j0..j0+31, row-major (4KB)
  __shared__ __bf16 pT[4][16][32];     // per-wave P tile (C-layout -> A-layout)

  // Q fragments (queries are the K matrix), K=64 split into 2 fragments
  v16bf q0 = *(const v16bf*)&Kb[(size_t)(m0 + lr) * HD + lh * 16];
  v16bf q1 = *(const v16bf*)&Kb[(size_t)(m0 + lr) * HD + 32 + lh * 16];

  float mst[8], lst[8];
  v8f o[4];
#pragma unroll
  for (int r = 0; r < 8; ++r) { mst[r] = -1e30f; lst[r] = 0.f; }
#pragma unroll
  for (int t = 0; t < 4; ++t) o[t] = (v8f){};

  for (int j0 = 0; j0 < Nseq; j0 += 32) {
    // ---- stage V tile (32 rows x 64 dims, 4KB) row-major into LDS ----
    {
      const int e0 = threadIdx.x * 16;  // 16 bf16 (32B) per thread
#if defined(HAVE_ASYNC_LDS)
      async_vec* gsrc = (async_vec*)&Vb[(size_t)j0 * HD + e0];
      async_vec* ldst = (async_vec*)(&vRow[0][0] + e0);
      __builtin_amdgcn_global_load_async_to_lds_b128(gsrc, ldst, 0, 0);
      __builtin_amdgcn_global_load_async_to_lds_b128(gsrc + 1, ldst + 1, 0, 0);
      __builtin_amdgcn_s_wait_asynccnt(0);
#else
      v16bf vv = *(const v16bf*)&Vb[(size_t)j0 * HD + e0];
      *(v16bf*)(&vRow[0][0] + e0) = vv;
#endif
    }
    __syncthreads();

    // ---- scores: two 16x16 tiles (cols j0..j0+31), K-reduction = 64 ----
    v16bf kb;
    v8f c_lo = (v8f){}, c_hi = (v8f){};
    kb = *(const v16bf*)&Kb[(size_t)(j0 + lr) * HD + lh * 16];
    c_lo = wmma_bf16(q0, kb, c_lo);
    kb = *(const v16bf*)&Kb[(size_t)(j0 + lr) * HD + 32 + lh * 16];
    c_lo = wmma_bf16(q1, kb, c_lo);
    kb = *(const v16bf*)&Kb[(size_t)(j0 + 16 + lr) * HD + lh * 16];
    c_hi = wmma_bf16(q0, kb, c_hi);
    kb = *(const v16bf*)&Kb[(size_t)(j0 + 16 + lr) * HD + 32 + lh * 16];
    c_hi = wmma_bf16(q1, kb, c_hi);

    // ---- online softmax update (per row r + 8*lh, 32 cols this step) ----
    float p_lo[8], p_hi[8], fac[8];
#pragma unroll
    for (int r = 0; r < 8; ++r) {
      int row = m0 + r + 8 * lh;
      float slo = c_lo[r] * 0.125f + bias[(size_t)row * Nseq + j0 + lr];
      float shi = c_hi[r] * 0.125f + bias[(size_t)row * Nseq + j0 + 16 + lr];
      float tmax = fmaxf(slo, shi);
#pragma unroll
      for (int msk = 1; msk < 16; msk <<= 1)
        tmax = fmaxf(tmax, __shfl_xor(tmax, msk, 32));
      float mnew = fmaxf(mst[r], tmax);
      float f = __expf(mst[r] - mnew);
      float plo = __expf(slo - mnew);
      float phi = __expf(shi - mnew);
      float ps = plo + phi;
#pragma unroll
      for (int msk = 1; msk < 16; msk <<= 1) ps += __shfl_xor(ps, msk, 32);
      lst[r] = lst[r] * f + ps;
      mst[r] = mnew;
      fac[r] = f;
      p_lo[r] = plo;
      p_hi[r] = phi;
    }
#pragma unroll
    for (int t = 0; t < 4; ++t)
#pragma unroll
      for (int r = 0; r < 8; ++r) o[t][r] *= fac[r];

    // ---- C-layout P -> bf16 A-layout via per-wave LDS region ----
#pragma unroll
    for (int r = 0; r < 8; ++r) {
      pT[w][r + 8 * lh][lr] = (__bf16)p_lo[r];
      pT[w][r + 8 * lh][lr + 16] = (__bf16)p_hi[r];
    }
    v16bf pa = *(const v16bf*)&pT[w][lr][lh * 16];

    // ---- out += P(16x32) @ V(32x64); B frags transpose-read from vRow ----
#pragma unroll
    for (int t = 0; t < 4; ++t) {
      v16bf vb;
#pragma unroll
      for (int i = 0; i < 16; ++i) vb[i] = vRow[lh * 16 + i][t * 16 + lr];
      o[t] = wmma_bf16(pa, vb, o[t]);
    }
    __syncthreads();
  }

  // ---- normalize and scatter to [B, N, C] intermediate ----
#pragma unroll
  for (int r = 0; r < 8; ++r) {
    float inv = 1.0f / lst[r];
    int m = m0 + r + 8 * lh;
#pragma unroll
    for (int t = 0; t < 4; ++t)
      aws[((size_t)b * Nseq + m) * Cdim + h * HD + t * 16 + lr] =
          (__bf16)(o[t][r] * inv);
  }
}

// ---------------------------------------------------------------------------
// Kernel 3: output projection.  out[m, j] = attn[m,:] . Wp[j,:] + bp[j]
// M = 8192, N = 768, K = 768.  Same tiling as kernel 1; f32 output.
// ---------------------------------------------------------------------------
__global__ __launch_bounds__(256) void proj_kernel(
    const __bf16* __restrict__ aws, const float* __restrict__ Wp,
    const float* __restrict__ bp, float* __restrict__ out) {
  const int lane = threadIdx.x & 31, w = threadIdx.x >> 5;
  const int lr = lane & 15, lh = lane >> 4;
  const int m0 = blockIdx.x * 128 + (w & 3) * 32;
  const int n0 = blockIdx.y * 128 + (w >> 2) * 64;

  v8f acc[2][4];
#pragma unroll
  for (int i = 0; i < 2; ++i)
#pragma unroll
    for (int t = 0; t < 4; ++t) acc[i][t] = (v8f){};

  for (int k0 = 0; k0 < Cdim; k0 += 32) {
    __builtin_prefetch(&aws[(size_t)(m0 + lr) * Cdim + k0 + 256], 0, 0);
    v16bf a[2], bm[4];
#pragma unroll
    for (int i = 0; i < 2; ++i)
      a[i] = *(const v16bf*)&aws[(size_t)(m0 + i * 16 + lr) * Cdim + k0 +
                                 lh * 16];
#pragma unroll
    for (int t = 0; t < 4; ++t)
      bm[t] = cvt16(&Wp[(size_t)(n0 + t * 16 + lr) * Cdim + k0 + lh * 16]);
#pragma unroll
    for (int i = 0; i < 2; ++i)
#pragma unroll
      for (int t = 0; t < 4; ++t) acc[i][t] = wmma_bf16(a[i], bm[t], acc[i][t]);
  }

#pragma unroll
  for (int i = 0; i < 2; ++i)
#pragma unroll
    for (int t = 0; t < 4; ++t)
#pragma unroll
      for (int r = 0; r < 8; ++r) {
        int m = m0 + i * 16 + r + 8 * lh;
        int j = n0 + t * 16 + lr;
        out[(size_t)m * Cdim + j] = acc[i][t][r] + bp[j];
      }
}

// ---------------------------------------------------------------------------
extern "C" void kernel_launch(void* const* d_in, const int* in_sizes, int n_in,
                              void* d_out, int out_size, void* d_ws,
                              size_t ws_size, hipStream_t stream) {
  const float* x    = (const float*)d_in[0];
  const float* bias = (const float*)d_in[1];
  const float* Wqkv = (const float*)d_in[2];
  const float* bqkv = (const float*)d_in[3];
  const float* Wp   = (const float*)d_in[4];
  const float* bp   = (const float*)d_in[5];
  float* out = (float*)d_out;

  const size_t kv_elems = (size_t)Bsz * Hn * Nseq * HD;  // 6,291,456
  char* ws = (char*)d_ws;
  __bf16* kws = (__bf16*)ws;
  __bf16* vws = (__bf16*)(ws + kv_elems * 2);
  __bf16* aws = (__bf16*)(ws + kv_elems * 4);
  // total workspace: 3 * kv_elems * 2 bytes = ~37.7 MB

  kv_proj_kernel<<<dim3(64, 12), 256, 0, stream>>>(x, Wqkv, bqkv, kws, vws);
  attn_kernel<<<dim3(Bsz * Hn * 16), 128, 0, stream>>>(kws, vws, bias, aws);
  proj_kernel<<<dim3(64, 6), 256, 0, stream>>>(aws, Wp, bp, out);
}
